// MetaAdaptiveFusion_48610439856449
// MI455X (gfx1250) — matrix operand order repaired and checked by reading, not compiled, not run on hardware
//
#include <hip/hip_runtime.h>
#include <hip/hip_bf16.h>
#include <math.h>

typedef __attribute__((ext_vector_type(16))) _Float16 v16h;
typedef __attribute__((ext_vector_type(8)))  _Float16 v8h;
typedef __attribute__((ext_vector_type(4)))  _Float16 v4h;
typedef __attribute__((ext_vector_type(8)))  float    v8f;

#define B_ 32
#define N_ 2048
#define H_ 256
#define F_ 8
#define NH_ 8
#define M_ 64   // F_*NH_ rows per batch

// ---------------------------------------------------------------------------
// Kernel 1: docs f32 -> f16, straight copy + LDS-tiled transposed copy.
// grid: 32 * 64 * 8 blocks (b, n-tile of 32, c-tile of 32), 256 threads.
// ---------------------------------------------------------------------------
__global__ __launch_bounds__(256) void conv_kernel(const float* __restrict__ docs,
                                                   _Float16* __restrict__ docs16,
                                                   _Float16* __restrict__ docsT16) {
  int bx  = blockIdx.x;
  int b   = bx >> 9;
  int rem = bx & 511;
  int tn  = rem >> 3;      // n tile 0..63
  int tc  = rem & 7;       // c tile 0..7
  int t   = threadIdx.x;
  int rl  = t >> 3;        // 0..31 row-in-tile (n)
  int c4  = (t & 7) * 4;   // 0..28 col offset (c)
  int n   = tn * 32 + rl;
  int c   = tc * 32 + c4;

  const float4 v = *(const float4*)(docs + ((size_t)b * N_ + n) * H_ + c);
  v4h h;
  h[0] = (_Float16)v.x; h[1] = (_Float16)v.y; h[2] = (_Float16)v.z; h[3] = (_Float16)v.w;
  *(v4h*)(docs16 + ((size_t)b * N_ + n) * H_ + c) = h;

  __shared__ _Float16 tile[32][33];
  tile[rl][c4 + 0] = h[0]; tile[rl][c4 + 1] = h[1];
  tile[rl][c4 + 2] = h[2]; tile[rl][c4 + 3] = h[3];
  __syncthreads();

  int cT = t >> 3;         // 0..31 c within tile
  int n4 = (t & 7) * 4;    // 0..28 n offset
  v4h o;
  o[0] = tile[n4 + 0][cT]; o[1] = tile[n4 + 1][cT];
  o[2] = tile[n4 + 2][cT]; o[3] = tile[n4 + 3][cT];
  *(v4h*)(docsT16 + ((size_t)b * H_ + tc * 32 + cT) * N_ + tn * 32 + n4) = o;
}

// ---------------------------------------------------------------------------
// Kernel 2: qh = query@WqT + bq ; qk = (qh @ Wk_slice) / sqrt(dh), f16 out.
// grid: 8*32 blocks (f,b), 256 threads. K-bias drops (softmax invariant).
// ---------------------------------------------------------------------------
__global__ __launch_bounds__(256) void qk_kernel(const float* __restrict__ q,
                                                 const float* __restrict__ Wi,
                                                 const float* __restrict__ bi,
                                                 _Float16* __restrict__ qk16) {
  int f = blockIdx.x >> 5, b = blockIdx.x & 31;
  int t = threadIdx.x;
  __shared__ float qs[H_];
  __shared__ float qh[H_];
  qs[t] = q[b * H_ + t];
  __syncthreads();
  const float* Wq = Wi + (size_t)f * 768 * H_;          // rows 0..255
  float acc = bi[f * 768 + t];
  for (int c = 0; c < H_; ++c) acc += qs[c] * Wq[t * H_ + c];
  qh[t] = acc;
  __syncthreads();
  const float* Wk = Wi + (size_t)f * 768 * H_ + 256 * H_;  // rows 256..511
  const float scale = 0.17677669529663687f;                 // 1/sqrt(32)
  for (int h = 0; h < NH_; ++h) {
    float s = 0.f;
    for (int d = 0; d < 32; ++d) s += qh[h * 32 + d] * Wk[(h * 32 + d) * H_ + t];
    qk16[((size_t)b * M_ + f * NH_ + h) * H_ + t] = (_Float16)(s * scale);
  }
}

// ---------------------------------------------------------------------------
// Kernel 3: GEMM1  scores[b,64,2048] = qk16[b,64,256] @ docs16[b,2048,256]^T
// grid: 32*8 blocks (b, 256-col strip), 8 waves. Wave: 16x128 tile, K=256.
// ---------------------------------------------------------------------------
__global__ __launch_bounds__(256) void gemm1_kernel(const _Float16* __restrict__ qk16,
                                                    const _Float16* __restrict__ docs16,
                                                    float* __restrict__ scores) {
  int b      = blockIdx.x >> 3;
  int nstrip = blockIdx.x & 7;
  int wave = threadIdx.x >> 5, lane = threadIdx.x & 31;
  int l16 = lane & 15, lh = lane >> 4;
  int mtile = wave & 3;
  int nhalf = wave >> 2;
  int ntile0 = nstrip * 16 + nhalf * 8;

  const _Float16* Arow = qk16 + ((size_t)b * M_ + mtile * 16 + l16) * H_ + lh * 16;
  const _Float16* Bb   = docs16 + (size_t)b * N_ * H_ + lh * 16;

  v8f acc[8] = {};
  for (int kk = 0; kk < H_; kk += 32) {
    v8h a0 = *(const v8h*)(Arow + kk);
    v8h a1 = *(const v8h*)(Arow + kk + 8);
    v16h A = __builtin_shufflevector(a0, a1, 0,1,2,3,4,5,6,7,8,9,10,11,12,13,14,15);
#pragma unroll
    for (int t = 0; t < 8; ++t) {
      v16h Bv = *(const v16h*)(Bb + (size_t)((ntile0 + t) * 16 + l16) * H_ + kk);
      acc[t] = __builtin_amdgcn_wmma_f32_16x16x32_f16(false, A, false, Bv,
                                                      (short)0, acc[t], false, false);
    }
  }
  float* Sb = scores + ((size_t)b * M_ + mtile * 16) * N_;
#pragma unroll
  for (int t = 0; t < 8; ++t) {
    int col = (ntile0 + t) * 16 + l16;
#pragma unroll
    for (int r = 0; r < 8; ++r) Sb[(size_t)(r + lh * 8) * N_ + col] = acc[t][r];
  }
}

// ---------------------------------------------------------------------------
// Kernel 4: softmax over n. scores *= w pre-softmax; attnw = attn*w (f16).
// grid: 2048 blocks (b*64+m), 256 threads (8 elems each).
// ---------------------------------------------------------------------------
__global__ __launch_bounds__(256) void softmax_kernel(const float* __restrict__ scores,
                                                      const float* __restrict__ cw,
                                                      _Float16* __restrict__ attnw16) {
  int row = blockIdx.x;
  int b = row >> 6;
  int t = threadIdx.x;
  const float* S = scores + (size_t)row * N_;
  const float* W = cw + (size_t)b * N_;
  float s[8], wv[8];
  float mx = -3.4e38f;
#pragma unroll
  for (int i = 0; i < 8; ++i) {
    int n = t + i * 256;
    wv[i] = W[n];
    s[i]  = S[n] * wv[i];
    mx = fmaxf(mx, s[i]);
  }
  __shared__ float red[256];
  red[t] = mx; __syncthreads();
  for (int off = 128; off; off >>= 1) { if (t < off) red[t] = fmaxf(red[t], red[t + off]); __syncthreads(); }
  mx = red[0]; __syncthreads();
  float sum = 0.f;
#pragma unroll
  for (int i = 0; i < 8; ++i) { s[i] = __expf(s[i] - mx); sum += s[i]; }
  red[t] = sum; __syncthreads();
  for (int off = 128; off; off >>= 1) { if (t < off) red[t] += red[t + off]; __syncthreads(); }
  float inv = 1.0f / red[0];
  _Float16* A = attnw16 + (size_t)row * N_;
#pragma unroll
  for (int i = 0; i < 8; ++i) A[t + i * 256] = (_Float16)(s[i] * inv * wv[i]);
}

// ---------------------------------------------------------------------------
// Kernel 5: GEMM2  wctx[b,64,256] = attnw16[b,64,2048] @ docs16[b,2048,256]
// B read from transposed copy docsT16[b,256,2048] -> contiguous K per lane.
// grid: 32*4*2 blocks (b, mtile, chalf), 8 waves, 1 ctile/wave, K=2048.
// ---------------------------------------------------------------------------
__global__ __launch_bounds__(256) void gemm2_kernel(const _Float16* __restrict__ attnw16,
                                                    const _Float16* __restrict__ docsT16,
                                                    float* __restrict__ wctx) {
  int bx = blockIdx.x;
  int b     = bx >> 3;
  int mtile = (bx >> 1) & 3;
  int chalf = bx & 1;
  int wave = threadIdx.x >> 5, lane = threadIdx.x & 31;
  int l16 = lane & 15, lh = lane >> 4;
  int ctile = chalf * 8 + wave;   // 0..15

  const _Float16* Arow = attnw16 + ((size_t)b * M_ + mtile * 16 + l16) * N_ + lh * 16;
  const _Float16* Bcol = docsT16 + ((size_t)b * H_ + ctile * 16 + l16) * N_ + lh * 16;

  v8f acc = {};
  for (int kk = 0; kk < N_; kk += 32) {
    v8h a0 = *(const v8h*)(Arow + kk);
    v8h a1 = *(const v8h*)(Arow + kk + 8);
    v16h A = __builtin_shufflevector(a0, a1, 0,1,2,3,4,5,6,7,8,9,10,11,12,13,14,15);
    v16h Bv = *(const v16h*)(Bcol + kk);
    acc = __builtin_amdgcn_wmma_f32_16x16x32_f16(false, A, false, Bv,
                                                 (short)0, acc, false, false);
  }
  float* Wb = wctx + ((size_t)b * M_ + mtile * 16) * H_;
  int col = ctile * 16 + l16;
#pragma unroll
  for (int r = 0; r < 8; ++r) Wb[(size_t)(r + lh * 8) * H_ + col] = acc[r];
}

// ---------------------------------------------------------------------------
// Kernel 6: ctx = Wv@wctx + bv (bias valid since softmax sums to 1),
//           head_out = ctx@WoT + bo.  grid: 8*32 (f,b), 256 threads.
// ---------------------------------------------------------------------------
__global__ __launch_bounds__(256) void headproj_kernel(const float* __restrict__ wctx,
                                                       const float* __restrict__ Wi,
                                                       const float* __restrict__ bi,
                                                       const float* __restrict__ Wo,
                                                       const float* __restrict__ bo,
                                                       float* __restrict__ hout) {
  int f = blockIdx.x >> 5, b = blockIdx.x & 31;
  int t = threadIdx.x;
  __shared__ float wc[NH_ * H_];
  for (int i = t; i < NH_ * H_; i += 256)
    wc[i] = wctx[((size_t)b * M_ + f * NH_) * H_ + i];
  __syncthreads();
  int h = t >> 5;
  const float* Wv = Wi + (size_t)f * 768 * H_ + 512 * H_;  // rows 512..767
  float acc = bi[f * 768 + 512 + t];
  const float* wrow = wc + h * H_;
  for (int c = 0; c < H_; ++c) acc += wrow[c] * Wv[t * H_ + c];
  __shared__ float ctx[H_];
  ctx[t] = acc; __syncthreads();
  const float* Wof = Wo + (size_t)f * H_ * H_;
  float o = bo[f * H_ + t];
  for (int j = 0; j < H_; ++j) o += ctx[j] * Wof[t * H_ + j];
  hout[((size_t)f * B_ + b) * H_ + t] = o;
}

// ---------------------------------------------------------------------------
// Kernel 7: one meta step y = x + x@M.  grid: 256 rows (f*32+b), 256 threads.
// ---------------------------------------------------------------------------
__global__ __launch_bounds__(256) void meta_kernel(const float* __restrict__ x,
                                                   const float* __restrict__ Mp,
                                                   float* __restrict__ y) {
  int r = blockIdx.x, t = threadIdx.x;
  __shared__ float xr[H_];
  xr[t] = x[(size_t)r * H_ + t];
  __syncthreads();
  float acc = xr[t];
  for (int j = 0; j < H_; ++j) acc += xr[j] * Mp[j * H_ + t];
  y[(size_t)r * H_ + t] = acc;
}

// ---------------------------------------------------------------------------
// Kernel 8: strategy softmax + weighted combine. grid: 32 (b), 256 threads.
// ---------------------------------------------------------------------------
__global__ __launch_bounds__(256) void combine_kernel(const float* __restrict__ hfin,
                                                      const float* __restrict__ q,
                                                      const float* __restrict__ sw,
                                                      const float* __restrict__ sb,
                                                      float* __restrict__ out) {
  int b = blockIdx.x, t = threadIdx.x;
  int lane = t & 31, f0 = t >> 5;    // warp f0 computes logit f0
  float p = 0.f;
  for (int c = lane; c < H_; c += 32) p += q[b * H_ + c] * sw[f0 * H_ + c];
#pragma unroll
  for (int off = 16; off; off >>= 1) p += __shfl_xor(p, off, 32);
  __shared__ float logits[F_];
  if (lane == 0) logits[f0] = p + sb[f0];
  __syncthreads();
  float mx = logits[0];
  for (int f = 1; f < F_; ++f) mx = fmaxf(mx, logits[f]);
  float e[F_], sum = 0.f;
  for (int f = 0; f < F_; ++f) { e[f] = __expf(logits[f] - mx); sum += e[f]; }
  float inv = 1.f / sum;
  float o = 0.f;
  for (int f = 0; f < F_; ++f) o += e[f] * inv * hfin[((size_t)f * B_ + b) * H_ + t];
  out[b * H_ + t] = o;
}

// ---------------------------------------------------------------------------
extern "C" void kernel_launch(void* const* d_in, const int* in_sizes, int n_in,
                              void* d_out, int out_size, void* d_ws, size_t ws_size,
                              hipStream_t stream) {
  const float* query   = (const float*)d_in[0];
  const float* docs    = (const float*)d_in[1];
  const float* cw      = (const float*)d_in[2];
  // d_in[3] context_history: unused by the reference
  const float* in_w    = (const float*)d_in[4];
  const float* in_b    = (const float*)d_in[5];
  const float* out_w   = (const float*)d_in[6];
  const float* out_b   = (const float*)d_in[7];
  const float* strat_w = (const float*)d_in[8];
  const float* strat_b = (const float*)d_in[9];
  const float* meta    = (const float*)d_in[10];
  float* out = (float*)d_out;

  char* w = (char*)d_ws;
  const size_t MB = 1ull << 20;
  _Float16* docs16  = (_Float16*)(w);              // 32 MB
  _Float16* docsT16 = (_Float16*)(w + 32 * MB);    // 32 MB
  _Float16* qk16    = (_Float16*)(w + 64 * MB);    //  1 MB
  _Float16* attnw16 = (_Float16*)(w + 65 * MB);    //  8 MB
  float*    scores  = (float*)   (w + 73 * MB);    // 16 MB
  float*    wctx    = (float*)   (w + 89 * MB);    //  2 MB
  float*    hA      = (float*)   (w + 91 * MB);    //  1 MB
  float*    hB      = (float*)   (w + 92 * MB);    //  1 MB

  conv_kernel<<<32 * 64 * 8, 256, 0, stream>>>(docs, docs16, docsT16);
  qk_kernel<<<F_ * B_, 256, 0, stream>>>(query, in_w, in_b, qk16);
  gemm1_kernel<<<B_ * 8, 256, 0, stream>>>(qk16, docs16, scores);
  softmax_kernel<<<B_ * M_, 256, 0, stream>>>(scores, cw, attnw16);
  gemm2_kernel<<<B_ * 4 * 2, 256, 0, stream>>>(attnw16, docsT16, wctx);
  headproj_kernel<<<F_ * B_, 256, 0, stream>>>(wctx, in_w, in_b, out_w, out_b, hA);
  meta_kernel<<<F_ * B_, 256, 0, stream>>>(hA, meta + 0 * H_ * H_, hB);
  meta_kernel<<<F_ * B_, 256, 0, stream>>>(hB, meta + 1 * H_ * H_, hA);
  meta_kernel<<<F_ * B_, 256, 0, stream>>>(hA, meta + 2 * H_ * H_, hB);
  combine_kernel<<<B_, 256, 0, stream>>>(hB, query, strat_w, strat_b, out);
}
